// Aggregator_79602923864073
// MI455X (gfx1250) — compile-verified, matrix operand order checked by reading, not run on hardware
//
#include <hip/hip_runtime.h>
#include <hip/hip_bf16.h>

#define N_USERS      100000
#define N_ITEMS       50000
#define N_ENTITIES   180000
#define N_USER_NODES 150000
#define DIM 64

typedef __attribute__((ext_vector_type(16))) _Float16 v16h;
typedef __attribute__((ext_vector_type(8)))  float    v8f;

// ---------------------------------------------------------------------------
// Scatter-sum: one wave (32 lanes) per edge; each lane handles a float2 of the
// 64-float row. sum[dst] += emb[src] * weight[rel]; cnt[dst] += 1.
// Accumulators live in d_ws (~125 MB total -> resident in MI455X's 192 MB L2,
// so the f32 atomics resolve on-chip).
// ---------------------------------------------------------------------------
__global__ void scatter_wsum_kernel(const float* __restrict__ emb,
                                    const float* __restrict__ weight,
                                    const int*   __restrict__ src_idx,
                                    const int*   __restrict__ dst_idx,
                                    const int*   __restrict__ etype,   // null -> relation 0
                                    float* __restrict__ sum,
                                    float* __restrict__ cnt,
                                    int n_edges) {
  const int lane = threadIdx.x & 31;
  const int edge = blockIdx.x * (blockDim.x >> 5) + (threadIdx.x >> 5);
  if (edge >= n_edges) return;

  const int s   = src_idx[edge];
  const int d   = dst_idx[edge];
  const int rel = etype ? etype[edge] : 0;

  const float2 e = ((const float2*)(emb    + (size_t)s   * DIM))[lane];
  const float2 w = ((const float2*)(weight + (size_t)rel * DIM))[lane];

  float* ds = sum + (size_t)d * DIM + lane * 2;
  atomicAdd(ds + 0, e.x * w.x);
  atomicAdd(ds + 1, e.y * w.y);
  if (lane == 0) atomicAdd(cnt + d, 1.0f);
}

// ---------------------------------------------------------------------------
// mean = sum / max(cnt,1), written in place; rows >= tail_start are also
// streamed straight into the output buffer (the concat tails).
// ---------------------------------------------------------------------------
__global__ void finalize_mean_kernel(float* __restrict__ sum,
                                     const float* __restrict__ cnt,
                                     int nrows,
                                     float* __restrict__ out_full,  // may be null
                                     int tail_start) {
  const size_t idx   = (size_t)blockIdx.x * blockDim.x + threadIdx.x;
  const size_t total = (size_t)nrows * DIM;
  if (idx >= total) return;
  const int row = (int)(idx >> 6);
  const float v = sum[idx] / fmaxf(cnt[row], 1.0f);
  sum[idx] = v;
  if (out_full && row >= tail_start) out_full[idx] = v;
}

// ---------------------------------------------------------------------------
// WMMA fragment loader: 16-bit A-matrix 16x32 layout (ISA 7.12.2).
// lane half h = lane>>4; elements 0..7 hold K = k0..k0+7, elements 8..15 hold
// K = k0+16..k0+23, with k0 = kBase + h*8. B = W^T uses the same loader since
// B[k][n] = W[colBase+n][k] is a contiguous row of W.
// ---------------------------------------------------------------------------
__device__ __forceinline__ v16h load_frag16(const float* __restrict__ rowp, int k0) {
  v16h v;
#pragma unroll
  for (int i = 0; i < 8; ++i) v[i] = (_Float16)rowp[k0 + i];
#pragma unroll
  for (int i = 0; i < 8; ++i) v[8 + i] = (_Float16)rowp[k0 + 16 + i];
  return v;
}

// ---------------------------------------------------------------------------
// g = sigmoid(XA @ WA^T + XB @ WB^T);  out = g*XA + (1-g)*XB
// One wave per 16x16 output tile; K=64 reduced with two v_wmma_f32_16x16x32_f16
// steps per GEMM (4 WMMAs per tile), epilogue fused in-register.
// nrows must be a multiple of 16 (50000 / 100000 both are).
// ---------------------------------------------------------------------------
__global__ void gate_fuse_wmma_kernel(const float* __restrict__ XA,
                                      const float* __restrict__ WA,
                                      const float* __restrict__ XB,
                                      const float* __restrict__ WB,
                                      float* __restrict__ out,
                                      int nrows) {
  const int lane = threadIdx.x & 31;
  const int wave = threadIdx.x >> 5;
  const int tile = blockIdx.x * (blockDim.x >> 5) + wave;
  const int ntiles = (nrows >> 4) * (DIM / 16);
  if (tile >= ntiles) return;                 // wave-uniform: EXEC stays all-ones

  const int rowBase = (tile >> 2) * 16;       // DIM/16 == 4 column tiles
  const int colBase = (tile & 3) * 16;
  const int m = lane & 15;
  const int h = lane >> 4;

  const float* xa_row = XA + (size_t)(rowBase + m) * DIM;
  const float* wa_row = WA + (size_t)(colBase + m) * DIM;
  const float* xb_row = XB + (size_t)(rowBase + m) * DIM;
  const float* wb_row = WB + (size_t)(colBase + m) * DIM;

  v8f c = {};
#pragma unroll
  for (int kk = 0; kk < DIM; kk += 32) {
    const int k0 = kk + h * 8;
    v16h aA = load_frag16(xa_row, k0);
    v16h bA = load_frag16(wa_row, k0);
    c = __builtin_amdgcn_wmma_f32_16x16x32_f16(false, aA, false, bA,
                                               (short)0, c, false, false);
    v16h aB = load_frag16(xb_row, k0);
    v16h bB = load_frag16(wb_row, k0);
    c = __builtin_amdgcn_wmma_f32_16x16x32_f16(false, aB, false, bB,
                                               (short)0, c, false, false);
  }

  // C/D layout: VGPR r -> row rowBase + r + 8*h; col = colBase + (lane&15)
  const int col = colBase + m;
#pragma unroll
  for (int r = 0; r < 8; ++r) {
    const int row = rowBase + r + 8 * h;
    const float g = 1.0f / (1.0f + __expf(-c[r]));
    const float a = XA[(size_t)row * DIM + col];
    const float b = XB[(size_t)row * DIM + col];
    out[(size_t)row * DIM + col] = g * a + (1.0f - g) * b;
  }
}

// ---------------------------------------------------------------------------
extern "C" void kernel_launch(void* const* d_in, const int* in_sizes, int n_in,
                              void* d_out, int out_size, void* d_ws, size_t ws_size,
                              hipStream_t stream) {
  const float* entity_emb = (const float*)d_in[0];
  const float* user_emb   = (const float*)d_in[1];
  const float* weight     = (const float*)d_in[2];
  const float* gate1_w    = (const float*)d_in[3];
  const float* gate2_w    = (const float*)d_in[4];
  const float* gate3_w    = (const float*)d_in[5];
  const int* edge_index   = (const int*)d_in[6];   // (2, E): [head | tail]
  const int* edge_type    = (const int*)d_in[7];
  const int* u_edge_index = (const int*)d_in[8];   // (2, UE)
  const int* u_edge_type  = (const int*)d_in[9];
  const int* mat_row      = (const int*)d_in[10];
  const int* mat_col      = (const int*)d_in[11];

  const int n_edges  = in_sizes[6] / 2;   // 1,500,000
  const int n_uedges = in_sizes[8] / 2;   // 1,000,000
  const int n_mat    = in_sizes[10];      // 1,500,000

  // Workspace layout (floats)
  float* ws       = (float*)d_ws;
  float* ent_sum  = ws;
  float* ent_cnt  = ent_sum + (size_t)N_ENTITIES * DIM;
  float* att_sum  = ent_cnt + N_ENTITIES;
  float* att_cnt  = att_sum + (size_t)N_USER_NODES * DIM;
  float* iu_sum   = att_cnt + N_USER_NODES;
  float* iu_cnt   = iu_sum + (size_t)N_ITEMS * DIM;
  float* ui_sum   = iu_cnt + N_ITEMS;
  float* ui_cnt   = ui_sum + (size_t)N_USERS * DIM;
  const size_t ws_floats = (size_t)(ui_cnt + N_USERS - ws);
  hipMemsetAsync(d_ws, 0, ws_floats * sizeof(float), stream);

  float* ent_out = (float*)d_out;                         // (180000, 64)
  float* usr_out = ent_out + (size_t)N_ENTITIES * DIM;    // (150000, 64)

  const int TPB = 256;
  const int EPB = TPB / 32;  // 8 edges per block (wave32)

  // entity_agg = scatter_mean(entity_emb[tail] * weight[etype], head)
  scatter_wsum_kernel<<<(n_edges + EPB - 1) / EPB, TPB, 0, stream>>>(
      entity_emb, weight, edge_index + n_edges, edge_index, edge_type,
      ent_sum, ent_cnt, n_edges);
  // attribute_agg = scatter_mean(user_emb[utail] * weight[uetype], uhead)
  scatter_wsum_kernel<<<(n_uedges + EPB - 1) / EPB, TPB, 0, stream>>>(
      user_emb, weight, u_edge_index + n_uedges, u_edge_index, u_edge_type,
      att_sum, att_cnt, n_uedges);
  // i_u_agg = scatter_mean(user_emb[mat_row] * weight[0], mat_col)
  scatter_wsum_kernel<<<(n_mat + EPB - 1) / EPB, TPB, 0, stream>>>(
      user_emb, weight, mat_row, mat_col, nullptr,
      iu_sum, iu_cnt, n_mat);
  // u_i_agg = scatter_mean(entity_emb[mat_col] * weight[0], mat_row)
  scatter_wsum_kernel<<<(n_mat + EPB - 1) / EPB, TPB, 0, stream>>>(
      entity_emb, weight, mat_col, mat_row, nullptr,
      ui_sum, ui_cnt, n_mat);

  // Finalize means; stream concat tails straight to d_out.
  {
    size_t tot = (size_t)N_ENTITIES * DIM;
    finalize_mean_kernel<<<(tot + TPB - 1) / TPB, TPB, 0, stream>>>(
        ent_sum, ent_cnt, N_ENTITIES, ent_out, N_ITEMS);
    tot = (size_t)N_USER_NODES * DIM;
    finalize_mean_kernel<<<(tot + TPB - 1) / TPB, TPB, 0, stream>>>(
        att_sum, att_cnt, N_USER_NODES, usr_out, N_USERS);
    tot = (size_t)N_ITEMS * DIM;
    finalize_mean_kernel<<<(tot + TPB - 1) / TPB, TPB, 0, stream>>>(
        iu_sum, iu_cnt, N_ITEMS, nullptr, N_ITEMS);
    tot = (size_t)N_USERS * DIM;
    finalize_mean_kernel<<<(tot + TPB - 1) / TPB, TPB, 0, stream>>>(
        ui_sum, ui_cnt, N_USERS, nullptr, N_USERS);
  }

  // Items: gi = sigmoid(kg@g1^T + iu@g2^T); out = gi*kg + (1-gi)*iu
  {
    const int ntiles = (N_ITEMS / 16) * (DIM / 16);  // 12500
    gate_fuse_wmma_kernel<<<(ntiles + EPB - 1) / EPB, TPB, 0, stream>>>(
        ent_sum /* item_kg_agg = entity_agg[:N_ITEMS] */, gate1_w,
        iu_sum, gate2_w, ent_out, N_ITEMS);
  }
  // Users: hi = sigmoid(ui@g2^T + ukg@g3^T); out = hi*ukg + (1-hi)*ui
  {
    const int ntiles = (N_USERS / 16) * (DIM / 16);  // 25000
    gate_fuse_wmma_kernel<<<(ntiles + EPB - 1) / EPB, TPB, 0, stream>>>(
        att_sum /* user_ukg_agg = attribute_agg[:N_USERS] */, gate3_w,
        ui_sum, gate2_w, usr_out, N_USERS);
  }
}
